// CharRNN_86792699118066
// MI455X (gfx1250) — matrix operand order repaired and checked by reading, not compiled, not run on hardware
//
#include <hip/hip_runtime.h>
#include <hip/hip_bf16.h>
#include <math.h>

// ---------------------------------------------------------------------------
// CharRNN on gfx1250 (MI455X): bf16 WMMA everywhere, f32 accumulate.
//   B=64, L=512, V=96, E=256, H=1024
// Kernels:
//   1) pack weights (W_e, W_h, fc_w^T) into hardware WMMA-B fragment layout,
//      convert embedding table to bf16
//   2) embed-gather + input projection  x_lin[l][b][h]  (whole-chip GEMM)
//   3) sequential recurrence: 4 WGs x 16 batch rows, h in LDS (double-buffered),
//      W_h streamed from L2, only workgroup barriers between timesteps
//   4) logits GEMM  (M=32768, N=96, K=1024) + bias
// ---------------------------------------------------------------------------

typedef __bf16 bf16;
typedef __attribute__((ext_vector_type(8)))  __bf16 v8bf;
typedef __attribute__((ext_vector_type(16))) __bf16 v16bf;
typedef __attribute__((ext_vector_type(8)))  float  v8f;

#define BATCH 64
#define LSEQ  512
#define VOCAB 96
#define EDIM  256
#define HDIM  1024
#define HPAD  1032   // LDS row stride in bf16 elems: 2064B -> rows land on distinct banks

__device__ __forceinline__ v16bf cat16(v8bf lo, v8bf hi) {
  return __builtin_shufflevector(lo, hi, 0,1,2,3,4,5,6,7,8,9,10,11,12,13,14,15);
}

// A-fragment (16x32 bf16): this lane's 16 values are at k%32 = {o..o+7, o+16..o+23},
// o = (lane>=16 ? 8 : 0).  p already points at row_base + kb*32 + o.
__device__ __forceinline__ v16bf load_a16(const bf16* p) {
  v8bf lo = *(const v8bf*)p;
  v8bf hi = *(const v8bf*)(p + 16);
  return cat16(lo, hi);
}

// B-fragment: pre-packed so this lane's 16 values are 32 contiguous bytes.
__device__ __forceinline__ v16bf load_b16(const bf16* p) {
  const v8bf* q = (const v8bf*)p;
  return cat16(q[0], q[1]);
}

__device__ __forceinline__ v8f wmma_bf16(v16bf a, v16bf b, v8f c) {
  return __builtin_amdgcn_wmma_f32_16x16x32_bf16(false, a, false, b, (short)0, c,
                                                 false, false);
}

// ---------------------------------------------------------------------------
// Kernel 1a: f32 -> bf16 convert (embedding table)
// ---------------------------------------------------------------------------
__global__ void cvt_bf16_kernel(const float* __restrict__ src,
                                bf16* __restrict__ dst, int n) {
  int i = blockIdx.x * blockDim.x + threadIdx.x;
  if (i < n) dst[i] = (bf16)src[i];
}

// ---------------------------------------------------------------------------
// Kernel 1b: pack a K x N matrix into WMMA-B fragment layout (bf16).
// Hardware B layout (16-bit, K=32 per tile): lane<16 holds col=lane, k=kb*32+{0..7,16..23};
// lane>=16 holds col=lane-16, k=kb*32+{8..15,24..31}.  Stored lane-contiguous:
//   dst[(((nb*KB + kb)*32) + lane)*16 + j]
// transposed=1: source is N x K row-major (fc_w -> B = fc_w^T).
// ---------------------------------------------------------------------------
__global__ void pack_b_kernel(const float* __restrict__ src, bf16* __restrict__ dst,
                              int K, int N, int KB, int total, int transposed) {
  int tid = blockIdx.x * blockDim.x + threadIdx.x;
  if (tid >= total) return;
  int j    = tid & 15;
  int lane = (tid >> 4) & 31;
  int blk  = tid >> 9;            // nb*KB + kb
  int kb   = blk % KB;
  int nb   = blk / KB;
  int k = kb * 32 + ((lane & 16) ? 8 : 0) + (j & 7) + ((j & 8) ? 16 : 0);
  int n = nb * 16 + (lane & 15);
  float v = transposed ? src[(size_t)n * K + k] : src[(size_t)k * N + n];
  dst[tid] = (bf16)v;
}

// ---------------------------------------------------------------------------
// Kernel 2: x_lin[l][b][h] = embedding[x[b][l]] @ W_e     (bf16 out, time-major)
// grid: 2048 blocks (512 l * 4 b-tiles), 256 threads (8 waves).
// wave w computes N columns [w*128, w*128+128) for one 16-token M tile.
// ---------------------------------------------------------------------------
__global__ void embed_proj_kernel(const long long* __restrict__ x,
                                  const bf16* __restrict__ emb,
                                  const bf16* __restrict__ wepack,
                                  bf16* __restrict__ xlin) {
  const int l      = blockIdx.x >> 2;
  const int b0     = (blockIdx.x & 3) * 16;
  const int lane   = threadIdx.x & 31;
  const int wave   = threadIdx.x >> 5;
  const int lanelo = lane & 15;
  const int o8     = (lane & 16) ? 8 : 0;
  const int KB     = EDIM / 32;   // 8

  // This lane's A rows are token (b0+lanelo, l)
  const int eidx = (int)x[(size_t)(b0 + lanelo) * LSEQ + l];
  const bf16* arow = emb + (size_t)eidx * EDIM;
  bf16* obase = xlin + ((size_t)l * BATCH + b0) * HDIM;

  for (int p = 0; p < 4; ++p) {
    const int nb0 = wave * 8 + p * 2;
    const int n0  = nb0 * 16 + lanelo;
    v8f acc0 = {}, acc1 = {};
    const bf16* bp0 = wepack + (size_t)(nb0 + 0) * KB * 512;
    const bf16* bp1 = wepack + (size_t)(nb0 + 1) * KB * 512;
#pragma unroll
    for (int kb = 0; kb < KB; ++kb) {
      v16bf a  = load_a16(arow + kb * 32 + o8);
      v16bf b0v = load_b16(bp0 + ((size_t)kb * 32 + lane) * 16);
      v16bf b1v = load_b16(bp1 + ((size_t)kb * 32 + lane) * 16);
      acc0 = wmma_bf16(a, b0v, acc0);
      acc1 = wmma_bf16(a, b1v, acc1);
    }
#pragma unroll
    for (int r = 0; r < 8; ++r) {
      int m = r + o8;   // o8 is 8 exactly when lane>=16
      obase[(size_t)m * HDIM + n0]      = (bf16)acc0[r];
      obase[(size_t)m * HDIM + n0 + 16] = (bf16)acc1[r];
    }
  }
}

// ---------------------------------------------------------------------------
// Kernel 3: recurrence.  grid=4 (16 batch rows each), 256 threads (8 waves).
// h lives in LDS (double buffered, bf16, padded rows); W_h streamed from L2
// straight into VGPRs (each B tile consumed exactly once per step per wave).
// Only an in-WGP barrier between timesteps (batch rows are independent).
// ---------------------------------------------------------------------------
__global__ void rnn_recurrence_kernel(const float* __restrict__ hidden0,
                                      const bf16* __restrict__ whpack,
                                      const bf16* __restrict__ xlin,
                                      bf16* __restrict__ hall,
                                      float* __restrict__ final_h) {
  extern __shared__ bf16 smem[];            // 2 * 16 * HPAD bf16 = 66048 B

  const int b0     = blockIdx.x * 16;
  const int lane   = threadIdx.x & 31;
  const int wave   = threadIdx.x >> 5;
  const int lanelo = lane & 15;
  const int o8     = (lane & 16) ? 8 : 0;
  const int KB     = HDIM / 32;             // 32

  // load initial hidden state (f32 -> bf16) into buffer 0
  for (int i = threadIdx.x; i < 16 * HDIM; i += blockDim.x) {
    int m = i >> 10, n = i & 1023;
    smem[m * HPAD + n] = (bf16)hidden0[(size_t)(b0 + m) * HDIM + n];
  }
  __syncthreads();

  int cur = 0;                               // 0 or 1: which half of smem holds h_t
  for (int t = 0; t < LSEQ; ++t) {
    bf16* hc = smem + cur * (16 * HPAD);     // current h (read)
    bf16* hn = smem + (cur ^ 1) * (16 * HPAD); // next h (write)
    const bf16* xbase = xlin + ((size_t)t * BATCH + b0) * HDIM;
    bf16*       obase = hall + ((size_t)t * BATCH + b0) * HDIM;
    const bf16* arow  = hc + lanelo * HPAD;

    for (int p = 0; p < 4; ++p) {
      const int nb0 = wave * 8 + p * 2;
      const int n0  = nb0 * 16 + lanelo;
      v8f acc0, acc1;
#pragma unroll
      for (int r = 0; r < 8; ++r) {         // init with x_lin (the "+ x" term)
        int m = r + o8;
        acc0[r] = (float)xbase[(size_t)m * HDIM + n0];
        acc1[r] = (float)xbase[(size_t)m * HDIM + n0 + 16];
      }
      const bf16* bp0 = whpack + (size_t)(nb0 + 0) * KB * 512;
      const bf16* bp1 = whpack + (size_t)(nb0 + 1) * KB * 512;
#pragma unroll 2
      for (int kb = 0; kb < KB; ++kb) {
        v16bf a   = load_a16(arow + kb * 32 + o8);
        v16bf b0v = load_b16(bp0 + ((size_t)kb * 32 + lane) * 16);
        v16bf b1v = load_b16(bp1 + ((size_t)kb * 32 + lane) * 16);
        acc0 = wmma_bf16(a, b0v, acc0);
        acc1 = wmma_bf16(a, b1v, acc1);
      }
#pragma unroll
      for (int r = 0; r < 8; ++r) {
        int m = r + o8;
        float v0 = tanhf(acc0[r]);
        float v1 = tanhf(acc1[r]);
        obase[(size_t)m * HDIM + n0]      = (bf16)v0;
        obase[(size_t)m * HDIM + n0 + 16] = (bf16)v1;
        hn[m * HPAD + n0]      = (bf16)v0;   // next step's A operand
        hn[m * HPAD + n0 + 16] = (bf16)v1;
        if (t == LSEQ - 1) {
          final_h[(size_t)(b0 + m) * HDIM + n0]      = v0;
          final_h[(size_t)(b0 + m) * HDIM + n0 + 16] = v1;
        }
      }
    }
    __syncthreads();   // reads of hc done + writes to hn visible
    cur ^= 1;
  }
}

// ---------------------------------------------------------------------------
// Kernel 4: logits[b][l][v] = h_all[b,l,:] @ fc_w^T + fc_b   (f32 out)
// M=32768 tokens, N=96 (exactly 6 N-tiles), K=1024.
// grid: 256 blocks x 8 waves; one 16-token M tile per wave, all 6 N-tiles.
// ---------------------------------------------------------------------------
__global__ void logits_kernel(const bf16* __restrict__ hall,
                              const bf16* __restrict__ fcpack,
                              const float* __restrict__ fc_b,
                              float* __restrict__ out) {
  const int mt     = blockIdx.x * 8 + (threadIdx.x >> 5);
  const int tok0   = mt * 16;
  const int b      = tok0 >> 9;     // tok / 512
  const int l0     = tok0 & 511;    // 16 consecutive tokens share b (16 | 512)
  const int lane   = threadIdx.x & 31;
  const int lanelo = lane & 15;
  const int o8     = (lane & 16) ? 8 : 0;
  const int KB     = HDIM / 32;     // 32

  // A row for this lane: token tok0+lanelo -> h_all[(l0+lanelo)*B + b][*]
  const bf16* arow = hall + ((size_t)(l0 + lanelo) * BATCH + b) * HDIM;

  v8f acc[6];
#pragma unroll
  for (int nt = 0; nt < 6; ++nt) {
    float bias = fc_b[nt * 16 + lanelo];
#pragma unroll
    for (int r = 0; r < 8; ++r) acc[nt][r] = bias;
  }
  for (int kb = 0; kb < KB; ++kb) {
    v16bf a = load_a16(arow + kb * 32 + o8);
#pragma unroll
    for (int nt = 0; nt < 6; ++nt) {
      v16bf bb = load_b16(fcpack + (((size_t)nt * KB + kb) * 32 + lane) * 16);
      acc[nt] = wmma_bf16(a, bb, acc[nt]);
    }
  }
#pragma unroll
  for (int nt = 0; nt < 6; ++nt) {
    int n = nt * 16 + lanelo;
#pragma unroll
    for (int r = 0; r < 8; ++r) {
      int m = r + o8;
      out[(size_t)(tok0 + m) * VOCAB + n] = acc[nt][r];
    }
  }
}

// ---------------------------------------------------------------------------
// Workspace layout (bytes), all 16B aligned:
//   emb_bf16 : [96][256]            @ 0         (49152 B)
//   W_e pack : 8 KB tiles           @ 49152     (524288 B)
//   W_h pack :                      @ 573440    (2097152 B)
//   fc_w pack:                      @ 2670592   (196608 B)
//   x_lin    : [512][64][1024] bf16 @ 2867200   (64 MB)
//   h_all    : [512][64][1024] bf16 @ 69976064  (64 MB)
// total ~131 MB
// ---------------------------------------------------------------------------
extern "C" void kernel_launch(void* const* d_in, const int* in_sizes, int n_in,
                              void* d_out, int out_size, void* d_ws, size_t ws_size,
                              hipStream_t stream) {
  const long long* x      = (const long long*)d_in[0];
  const float*     hidden = (const float*)d_in[1];
  const float*     emb    = (const float*)d_in[2];
  const float*     We     = (const float*)d_in[3];
  const float*     Wh     = (const float*)d_in[4];
  const float*     fcw    = (const float*)d_in[5];
  const float*     fcb    = (const float*)d_in[6];
  float* out = (float*)d_out;

  char* ws = (char*)d_ws;
  bf16* embb = (bf16*)(ws + 0);
  bf16* wep  = (bf16*)(ws + 49152);
  bf16* whp  = (bf16*)(ws + 573440);
  bf16* fcp  = (bf16*)(ws + 2670592);
  bf16* xlin = (bf16*)(ws + 2867200);
  bf16* hall = (bf16*)(ws + 69976064);

  // --- prep: convert + pack weights ---
  cvt_bf16_kernel<<<(VOCAB * EDIM + 255) / 256, 256, 0, stream>>>(emb, embb,
                                                                  VOCAB * EDIM);
  pack_b_kernel<<<262144 / 256, 256, 0, stream>>>(We, wep, EDIM, HDIM,
                                                  EDIM / 32, 262144, 0);
  pack_b_kernel<<<1048576 / 256, 256, 0, stream>>>(Wh, whp, HDIM, HDIM,
                                                   HDIM / 32, 1048576, 0);
  pack_b_kernel<<<98304 / 256, 256, 0, stream>>>(fcw, fcp, HDIM, VOCAB,
                                                 HDIM / 32, 98304, 1);

  // --- input projection (whole-chip WMMA GEMM) ---
  embed_proj_kernel<<<2048, 256, 0, stream>>>(x, embb, wep, xlin);

  // --- sequential recurrence (4 independent batch-tiles) ---
  const size_t lds_bytes = 2u * 16u * HPAD * sizeof(bf16);   // 66048
  rnn_recurrence_kernel<<<4, 256, lds_bytes, stream>>>(
      hidden, whp, xlin, hall, out + (size_t)BATCH * LSEQ * VOCAB);

  // --- logits GEMM + bias ---
  logits_kernel<<<256, 256, 0, stream>>>(hall, fcp, fcb, out);
}
